// GNN_19378892439961
// MI455X (gfx1250) — matrix operand order repaired and checked by reading, not compiled, not run on hardware
//
#include <hip/hip_runtime.h>

// ---------------------------------------------------------------------------
// 2-layer GCN for MI455X (gfx1250, wave32).
//  - GEMMs: v_wmma_f32_16x16x32_bf16, fully unrolled K loop (templated dims),
//    16xK A panel staged once in LDS (bf16, relu fused), W pre-packed into
//    fragment-major bf16 layout so each lane loads its B fragment as one
//    aligned 32B chunk (2x global_load_b128).
//  - Edge scatter: one block per edge, float atomics, global prefetch of the
//    gathered source row.
// ---------------------------------------------------------------------------

typedef __attribute__((ext_vector_type(16))) __bf16 v16bf;
typedef __attribute__((ext_vector_type(8)))  float  v8f;

#define IN_DIM   128
#define HID      256
#define OUT_DIM  128
#define N_GRAPHS 64

// ---------------- small utility kernels ----------------

__global__ void k_fill(float* p, size_t n, float v) {
    size_t i = (size_t)blockIdx.x * blockDim.x + threadIdx.x;
    if (i < n) p[i] = v;
}

__global__ void k_deg_count(const int* __restrict__ dst, float* __restrict__ deg, int nE) {
    int i = blockIdx.x * blockDim.x + threadIdx.x;
    if (i < nE) atomicAdd(&deg[dst[i]], 1.0f);
}

__global__ void k_rsqrt_inplace(float* p, int n) {
    int i = blockIdx.x * blockDim.x + threadIdx.x;
    if (i < n) {
        float d = p[i];
        p[i] = (d > 0.0f) ? rsqrtf(d) : 0.0f;
    }
}

// Pack W[K,F] (f32 row-major) into WMMA B-fragment-major bf16:
//   o[(kb*(F/16) + ct)*512 + lane*16 + elem]
// where per ISA B layout: half=lane>>4 selects K range, r=lane&15 is the
// column, elem maps to K = kb*32 + (elem<8 ? 0 : 16) + half*8 + (elem&7).
template<int K, int F>
__global__ void k_pack_w(const float* __restrict__ w, __bf16* __restrict__ o) {
    int i = blockIdx.x * blockDim.x + threadIdx.x;
    if (i >= K * F) return;
    int elem = i & 15;
    int lane = (i >> 4) & 31;
    int ct   = (i >> 9) % (F / 16);
    int kb   = i / (512 * (F / 16));
    int half = lane >> 4, r = lane & 15;
    int j = elem >> 1, odd = elem & 1;
    int kk  = kb * 32 + ((elem < 8) ? 0 : 16) + half * 8 + (elem & 7);
    int col = ct * 16 + r;
    o[i] = (__bf16)w[kk * F + col];
}

// ---------------- WMMA bf16 GEMM:  out[N,F] = (relu?)A[N,K] @ W[K,F] --------
// grid = (N/16, F/128), block = 256 (8 waves). Requires N % 16 == 0.
// Each wave computes one 16x16 tile; all 8 waves share one 16xK LDS A panel.

template<int K, int F, bool RELU>
__global__ __launch_bounds__(256)
void k_gemm_bf16(const float* __restrict__ A, const __bf16* __restrict__ Wpk,
                 float* __restrict__ out)
{
    constexpr int LDK = K + 8;                    // pad to dodge bank conflicts
    __shared__ __bf16 ldsA[16 * LDK];

    const int tid  = threadIdx.x;
    const int wave = tid >> 5;
    const int lane = tid & 31;
    const int half = lane >> 4;                   // lane group 0 / 1
    const int r    = lane & 15;
    const int row0 = blockIdx.x * 16;
    const int colTile = blockIdx.y * 8 + wave;    // 16-col tile index
    const int col  = colTile * 16 + r;

    // stage the full 16xK A panel as bf16 (relu fused), coalesced f32 reads
    #pragma unroll
    for (int idx = tid; idx < 16 * K; idx += 256) {
        int rr = idx / K, kk = idx % K;
        float v = A[(size_t)(row0 + rr) * K + kk];
        if (RELU) v = fmaxf(v, 0.0f);
        ldsA[rr * LDK + kk] = (__bf16)v;
    }
    __syncthreads();

    v8f c = {};
    #pragma unroll
    for (int kb = 0; kb < K / 32; ++kb) {
        // A fragment per ISA 16-bit A layout (from LDS)
        v16bf a;
        #pragma unroll
        for (int j = 0; j < 8; ++j) {
            int kbase = kb * 32 + ((j < 4) ? 0 : 16) + half * 8 + ((j & 3) << 1);
            a[2 * j]     = ldsA[r * LDK + kbase];
            a[2 * j + 1] = ldsA[r * LDK + kbase + 1];
        }
        // B fragment: one contiguous, 32B-aligned chunk per lane
        v16bf b = *(const v16bf*)(Wpk +
                   ((size_t)(kb * (F / 16) + colTile) * 32 + lane) * 16);

        c = __builtin_amdgcn_wmma_f32_16x16x32_bf16(
                /*neg_a=*/false, a, /*neg_b=*/false, b,
                /*c_mod=*/(short)0, c, /*reuse_a=*/false, /*reuse_b=*/false);
    }

    // C/D layout: lane = column, rows half*8 + v
    float* op = out + (size_t)(row0 + half * 8) * F + col;
    #pragma unroll
    for (int v = 0; v < 8; ++v)
        op[(size_t)v * F] = c[v];
}

// ---------------- scatter stage ----------------

// h[n,f] = xl[n,f] * dinv[n]^2 + bias[f]   (self-loop term + bias; also init)
__global__ void k_self_bias(const float* __restrict__ xl, const float* __restrict__ bias,
                            const float* __restrict__ dinv, float* __restrict__ h,
                            int N, int F)
{
    size_t i = (size_t)blockIdx.x * blockDim.x + threadIdx.x;
    if (i >= (size_t)N * F) return;
    int n = (int)(i / F);
    int f = (int)(i % F);
    float di = dinv[n];
    h[i] = xl[i] * di * di + bias[f];
}

// h[dst] += xl[src] * dinv[src]*dinv[dst]  -- one block per edge
__global__ void k_edge_scatter(const float* __restrict__ xl,
                               const int* __restrict__ src, const int* __restrict__ dst,
                               const float* __restrict__ dinv, float* __restrict__ h,
                               int F, int nE)
{
    int e = blockIdx.x;
    if (e >= nE) return;
    int s = src[e], d = dst[e];
    float nrm = dinv[s] * dinv[d];
    const float* xs = xl + (size_t)s * F;
    float*       hd = h  + (size_t)d * F;
    __builtin_prefetch(xs, 0, 0);                 // global_prefetch_b8
    for (int f = threadIdx.x; f < F; f += blockDim.x)
        atomicAdd(&hd[f], xs[f] * nrm);
}

// ---------------- mean pool ----------------

__global__ void k_pool_accum(const float* __restrict__ h, const int* __restrict__ batch,
                             float* __restrict__ sums, float* __restrict__ cnt, int N)
{
    int n = blockIdx.x;
    if (n >= N) return;
    int g = batch[n];
    if (threadIdx.x == 0) atomicAdd(&cnt[g], 1.0f);
    atomicAdd(&sums[(size_t)g * OUT_DIM + threadIdx.x],
              h[(size_t)n * OUT_DIM + threadIdx.x]);
}

__global__ void k_pool_out(const float* __restrict__ sums, const float* __restrict__ cnt,
                           float* __restrict__ out)
{
    int i = blockIdx.x * blockDim.x + threadIdx.x;
    if (i >= N_GRAPHS * OUT_DIM) return;
    int g = i / OUT_DIM;
    out[i] = sums[i] / fmaxf(cnt[g], 1.0f);
}

// ---------------- launcher ----------------

extern "C" void kernel_launch(void* const* d_in, const int* in_sizes, int n_in,
                              void* d_out, int out_size, void* d_ws, size_t ws_size,
                              hipStream_t stream)
{
    const float* x     = (const float*)d_in[0];
    const int*   eidx  = (const int*)d_in[1];     // [2, E]
    const int*   batch = (const int*)d_in[2];
    const float* W1    = (const float*)d_in[3];
    const float* b1    = (const float*)d_in[4];
    const float* W2    = (const float*)d_in[5];
    const float* b2    = (const float*)d_in[6];
    float*       out   = (float*)d_out;

    const int N = in_sizes[0] / IN_DIM;           // 100000 (divisible by 16)
    const int E = in_sizes[1] / 2;                // 1600000
    const int* src = eidx;
    const int* dst = eidx + E;

    // workspace carve-out (256B aligned)
    char*  ws  = (char*)d_ws;
    size_t off = 0;
    auto carve = [&](size_t bytes) {
        size_t o = off;
        off = (off + bytes + 255) & ~(size_t)255;
        return (void*)(ws + o);
    };
    __bf16* w1pk = (__bf16*)carve((size_t)IN_DIM * HID * sizeof(__bf16));
    __bf16* w2pk = (__bf16*)carve((size_t)HID * OUT_DIM * sizeof(__bf16));
    float*  dinv = (float*)carve((size_t)N * sizeof(float));
    float*  xl   = (float*)carve((size_t)N * HID * sizeof(float)); // xl1 then xl2
    float*  h    = (float*)carve((size_t)N * HID * sizeof(float)); // h1 then h2
    float*  sums = (float*)carve((size_t)N_GRAPHS * OUT_DIM * sizeof(float));
    float*  cnt  = (float*)carve((size_t)N_GRAPHS * sizeof(float));
    (void)ws_size; (void)n_in; (void)out_size;

    // degrees -> dinv (self-loop contributes 1 to every node)
    k_fill<<<(N + 255) / 256, 256, 0, stream>>>(dinv, (size_t)N, 1.0f);
    k_deg_count<<<(E + 255) / 256, 256, 0, stream>>>(dst, dinv, E);
    k_rsqrt_inplace<<<(N + 255) / 256, 256, 0, stream>>>(dinv, N);

    // fragment-major bf16 weight packs
    k_pack_w<IN_DIM, HID><<<(IN_DIM * HID + 255) / 256, 256, 0, stream>>>(W1, w1pk);
    k_pack_w<HID, OUT_DIM><<<(HID * OUT_DIM + 255) / 256, 256, 0, stream>>>(W2, w2pk);

    // ---- layer 1 ----
    {
        dim3 grid(N / 16, HID / 128);
        k_gemm_bf16<IN_DIM, HID, false><<<grid, 256, 0, stream>>>(x, w1pk, xl);
    }
    k_self_bias<<<(int)(((size_t)N * HID + 255) / 256), 256, 0, stream>>>(xl, b1, dinv, h, N, HID);
    k_edge_scatter<<<E, 256, 0, stream>>>(xl, src, dst, dinv, h, HID, E);

    // ---- layer 2 (relu fused into GEMM A-load) ----
    {
        dim3 grid(N / 16, OUT_DIM / 128);
        k_gemm_bf16<HID, OUT_DIM, true><<<grid, 256, 0, stream>>>(h, w2pk, xl);
    }
    k_self_bias<<<(int)(((size_t)N * OUT_DIM + 255) / 256), 256, 0, stream>>>(xl, b2, dinv, h, N, OUT_DIM);
    k_edge_scatter<<<E, 128, 0, stream>>>(xl, src, dst, dinv, h, OUT_DIM, E);

    // ---- mean pool ----
    k_fill<<<(N_GRAPHS * OUT_DIM + N_GRAPHS + 255) / 256, 256, 0, stream>>>(
        sums, (size_t)(N_GRAPHS * OUT_DIM + N_GRAPHS), 0.0f);   // sums + cnt (contiguous)
    k_pool_accum<<<N, OUT_DIM, 0, stream>>>(h, batch, sums, cnt, N);
    k_pool_out<<<(N_GRAPHS * OUT_DIM + 255) / 256, 256, 0, stream>>>(sums, cnt, out);
}
